// Attention_15702400434534
// MI455X (gfx1250) — matrix operand order repaired and checked by reading, not compiled, not run on hardware
//
#include <hip/hip_runtime.h>

// ---------------------------------------------------------------------------
// Problem constants (from the reference)
// ---------------------------------------------------------------------------
#define S_TOT   2048
#define DIM     4096
#define NH      32
#define NKV     8
#define HD      128
#define QCOLS   (NH * HD)    // 4096
#define KVCOLS  (NKV * HD)   // 1024
#define WIN     1024

typedef __bf16 bf16;
typedef bf16  v16bf __attribute__((ext_vector_type(16)));
typedef float v8f   __attribute__((ext_vector_type(8)));

static __device__ __forceinline__ bf16 f2b(float f) { return (bf16)f; }

// LDS byte offset of a shared-memory pointer (LDS aperture keeps the
// workgroup-relative offset in the low 32 bits of the flat address).
static __device__ __forceinline__ uint32_t lds_off(const void* p) {
    return (uint32_t)(size_t)p;
}

// CDNA5 async copy: global -> LDS, 16B per lane, tracked by ASYNCcnt.
static __device__ __forceinline__ void async_ld_b128(uint32_t dst_lds, const void* src) {
    asm volatile("global_load_async_to_lds_b128 %0, %1, off"
                 :: "v"(dst_lds), "v"(src) : "memory");
}
static __device__ __forceinline__ void wait_async0() {
    asm volatile("s_wait_asynccnt 0x0" ::: "memory");
}

// ---------------------------------------------------------------------------
// fp32 -> bf16 conversion (layout preserved)
// ---------------------------------------------------------------------------
__global__ __launch_bounds__(256) void cvt_f32_bf16(const float* __restrict__ src,
                                                    bf16* __restrict__ dst, int n) {
    int i = blockIdx.x * 256 + threadIdx.x;
    if (i < n) dst[i] = f2b(src[i]);
}

// ---------------------------------------------------------------------------
// fp32 [R][C] -> bf16 [C][R] transpose-convert (32x32 LDS tiles)
// ---------------------------------------------------------------------------
__global__ __launch_bounds__(256) void cvt_transpose_bf16(const float* __restrict__ src,
                                                          bf16* __restrict__ dst,
                                                          int R, int C) {
    __shared__ float tile[32][33];
    const int rb = blockIdx.y * 32, cb = blockIdx.x * 32;
    const int tr = threadIdx.x >> 5;
    const int tc = threadIdx.x & 31;
    #pragma unroll
    for (int i = 0; i < 4; ++i)
        tile[tr + i * 8][tc] = src[(size_t)(rb + tr + i * 8) * C + cb + tc];
    __syncthreads();
    #pragma unroll
    for (int i = 0; i < 4; ++i)
        dst[(size_t)(cb + tr + i * 8) * R + rb + tc] = f2b(tile[tc][tr + i * 8]);
}

// ---------------------------------------------------------------------------
// RoPE (interleaved pairs, per reference) + bf16 conversion
// ---------------------------------------------------------------------------
__global__ __launch_bounds__(256) void rope_bf16(const float* __restrict__ in,
                                                 const float* __restrict__ cosb,
                                                 const float* __restrict__ sinb,
                                                 bf16* __restrict__ out, int nheads) {
    int i = blockIdx.x * 256 + threadIdx.x;
    int total = S_TOT * nheads * (HD / 2);
    if (i >= total) return;
    int p = i & 63;
    int t = i >> 6;
    int s = t / nheads;
    float tr = in[t * HD + 2 * p];
    float ti = in[t * HD + 2 * p + 1];
    float c  = cosb[s * 64 + p];
    float sn = sinb[s * 64 + p];
    out[t * HD + 2 * p]     = f2b(tr * c - ti * sn);
    out[t * HD + 2 * p + 1] = f2b(tr * sn + ti * c);
}

// ---------------------------------------------------------------------------
// bf16 WMMA GEMM:  C[M][N] f32 = A[M][K] @ Bt[N][K] (B pre-transposed)
// 128x128 block tile, K-step 32, 8 waves x (32x64).
// Staging uses CDNA5 global_load_async_to_lds_b128 with double-buffered LDS:
// one barrier per iteration, async copies overlap the 8 WMMAs.
// ---------------------------------------------------------------------------
__global__ __launch_bounds__(256) void gemm_bf16_wmma(const bf16* __restrict__ A,
                                                      const bf16* __restrict__ Bt,
                                                      float* __restrict__ C,
                                                      int M, int N, int K) {
    __shared__ alignas(16) bf16 As[2][128 * 40];
    __shared__ alignas(16) bf16 Bs[2][128 * 40];

    const int tid   = threadIdx.x;
    const int wave  = tid >> 5;
    const int lane  = tid & 31;
    const int lhalf = lane >> 4;
    const int l15   = lane & 15;

    const int bm = blockIdx.y * 128;
    const int bn = blockIdx.x * 128;
    const int wm = (wave & 3) * 32;
    const int wn = (wave >> 2) * 64;

    // staging coordinates: idx = tid + it*256, row = idx>>2, col8 = (idx&3)*8
    const int r0 = tid >> 2;
    const int r1 = (tid + 256) >> 2;
    const int cc = (tid & 3) * 8;

    const bf16* gA0 = A  + (size_t)(bm + r0) * K + cc;
    const bf16* gA1 = A  + (size_t)(bm + r1) * K + cc;
    const bf16* gB0 = Bt + (size_t)(bn + r0) * K + cc;
    const bf16* gB1 = Bt + (size_t)(bn + r1) * K + cc;

    const uint32_t oA0 = lds_off(&As[0][r0 * 40 + cc]);
    const uint32_t oA1 = lds_off(&As[0][r1 * 40 + cc]);
    const uint32_t oB0 = lds_off(&Bs[0][r0 * 40 + cc]);
    const uint32_t oB1 = lds_off(&Bs[0][r1 * 40 + cc]);
    const uint32_t bufB = 128 * 40 * 2;   // bytes per buffer

    v8f acc[2][4] = {};

    // prologue: async-stage K-tile 0 into buffer 0
    async_ld_b128(oA0, gA0);
    async_ld_b128(oA1, gA1);
    async_ld_b128(oB0, gB0);
    async_ld_b128(oB1, gB1);

    int buf = 0;
    for (int k0 = 0; k0 < K; k0 += 32, buf ^= 1) {
        wait_async0();       // my copies into LDS[buf] are complete
        __syncthreads();     // everyone's are

        if (k0 + 32 < K) {   // async-stage next tile into LDS[buf^1]
            uint32_t bo = (uint32_t)(buf ^ 1) * bufB;
            async_ld_b128(oA0 + bo, gA0 + k0 + 32);
            async_ld_b128(oA1 + bo, gA1 + k0 + 32);
            async_ld_b128(oB0 + bo, gB0 + k0 + 32);
            async_ld_b128(oB1 + bo, gB1 + k0 + 32);
        }

        // fragments (two ds_load_b128 each, matching ISA VGPR layouts)
        v16bf afrag[2], bfrag[4];
        #pragma unroll
        for (int i = 0; i < 2; ++i) {
            const bf16* p = As[buf] + (wm + i * 16 + l15) * 40 + lhalf * 8;
            uint4* u = (uint4*)&afrag[i];
            u[0] = *(const uint4*)p;
            u[1] = *(const uint4*)(p + 16);
        }
        #pragma unroll
        for (int j = 0; j < 4; ++j) {
            const bf16* p = Bs[buf] + (wn + j * 16 + l15) * 40 + lhalf * 8;
            uint4* u = (uint4*)&bfrag[j];
            u[0] = *(const uint4*)p;
            u[1] = *(const uint4*)(p + 16);
        }
        #pragma unroll
        for (int i = 0; i < 2; ++i)
            #pragma unroll
            for (int j = 0; j < 4; ++j)
                acc[i][j] = __builtin_amdgcn_wmma_f32_16x16x32_bf16(
                    false, afrag[i], false, bfrag[j], (short)0, acc[i][j], false, false);
    }

    #pragma unroll
    for (int i = 0; i < 2; ++i)
        #pragma unroll
        for (int j = 0; j < 4; ++j) {
            int row0 = bm + wm + i * 16 + lhalf * 8;
            int col  = bn + wn + j * 16 + l15;
            #pragma unroll
            for (int r = 0; r < 8; ++r)
                C[(size_t)(row0 + r) * N + col] = acc[i][j][r];
        }
}

// ---------------------------------------------------------------------------
// Flash attention: block = (head, 128 q rows), 8 waves x 16 rows.
// K/V blocks async-copied to LDS (K natural-major, V pre-transposed in global).
// Analytic mask (seqs 1024/512/512, causal, window 1024). P transposed
// C->A layout via per-wave LDS with s_wait_dscnt.
// ---------------------------------------------------------------------------
__global__ __launch_bounds__(256) void attn_flash_wmma(const bf16* __restrict__ Q,
                                                       const bf16* __restrict__ Kc,
                                                       const bf16* __restrict__ Vt,
                                                       bf16* __restrict__ O) {
    __shared__ alignas(16) bf16 Ks[32 * 136];   // [key][dim]
    __shared__ alignas(16) bf16 Vs[128 * 40];   // [dim][key]
    __shared__ alignas(16) bf16 Ps[8][16 * 40]; // per-wave P scratch [row][key]

    const int h   = blockIdx.x;
    const int q0  = blockIdx.y * 128;
    const int kvh = h >> 2;
    const int tid = threadIdx.x;
    const int wave  = tid >> 5;
    const int lane  = tid & 31;
    const int lhalf = lane >> 4;
    const int l15   = lane & 15;
    const int m0 = q0 + wave * 16;

    const int sb = (q0 >= 1536) ? 1536 : ((q0 >= 1024) ? 1024 : 0);
    const int kstart  = max(sb, q0 - (WIN - 1)) & ~31;
    const int kend    = q0 + 127;
    const int wv_kmin = max(sb, m0 - (WIN - 1));
    const int wv_kmax = m0 + 15;

    // K staging coords: 32 keys x 128 dims in two rounds of 256x16B
    const int kl0 = tid >> 4,  kl1 = (tid + 256) >> 4;
    const int kd0 = (tid & 15) * 8;
    const uint32_t oK0 = lds_off(&Ks[kl0 * 136 + kd0]);
    const uint32_t oK1 = lds_off(&Ks[kl1 * 136 + kd0]);
    // V staging coords (Vt is [kvh*HD+d][s]): 128 dims x 32 keys
    const int vd0 = tid >> 2,  vd1 = (tid + 256) >> 2;
    const int vc8 = (tid & 3) * 8;
    const uint32_t oV0 = lds_off(&Vs[vd0 * 40 + vc8]);
    const uint32_t oV1 = lds_off(&Vs[vd1 * 40 + vc8]);
    const bf16* vbase0 = Vt + (size_t)(kvh * HD + vd0) * S_TOT + vc8;
    const bf16* vbase1 = Vt + (size_t)(kvh * HD + vd1) * S_TOT + vc8;

    // Q fragments: 4 x (16x32) in A layout, straight from global
    v16bf qf[4];
    {
        const bf16* qbase = Q + ((size_t)(m0 + l15) * NH + h) * HD;
        #pragma unroll
        for (int f = 0; f < 4; ++f) {
            const bf16* p = qbase + f * 32 + lhalf * 8;
            uint4* u = (uint4*)&qf[f];
            u[0] = *(const uint4*)p;
            u[1] = *(const uint4*)(p + 16);
        }
    }

    float mrow[8], lrow[8];
    v8f acc[8] = {};
    #pragma unroll
    for (int r = 0; r < 8; ++r) { mrow[r] = -3e38f; lrow[r] = 0.f; }
    const float scale = 0.088388347648318447f;   // 1/sqrt(128)

    for (int kk = kstart; kk <= kend; kk += 32) {
        __syncthreads();   // previous block fully consumed by all waves
        {
            const bf16* g0 = Kc + ((size_t)(kk + kl0) * NKV + kvh) * HD + kd0;
            const bf16* g1 = Kc + ((size_t)(kk + kl1) * NKV + kvh) * HD + kd0;
            async_ld_b128(oK0, g0);
            async_ld_b128(oK1, g1);
            async_ld_b128(oV0, vbase0 + kk);
            async_ld_b128(oV1, vbase1 + kk);
            if (kk + 32 <= kend) {
                __builtin_prefetch(g0 + (size_t)32 * NKV * HD, 0, 1);
                __builtin_prefetch(vbase0 + kk + 32, 0, 1);
            }
        }
        wait_async0();
        __syncthreads();   // K/V block visible to all waves

        if (kk + 31 < wv_kmin || kk > wv_kmax) continue;

        // scores: two 16(q) x 16(key) tiles over HD=128
        v8f sc[2];
        #pragma unroll
        for (int ncol = 0; ncol < 2; ++ncol) {
            v8f s = {};
            #pragma unroll
            for (int f = 0; f < 4; ++f) {
                v16bf kfrag;
                const bf16* p = Ks + (ncol * 16 + l15) * 136 + f * 32 + lhalf * 8;
                uint4* u = (uint4*)&kfrag;
                u[0] = *(const uint4*)p;
                u[1] = *(const uint4*)(p + 16);
                s = __builtin_amdgcn_wmma_f32_16x16x32_bf16(
                    false, qf[f], false, kfrag, (short)0, s, false, false);
            }
            sc[ncol] = s;
        }

        // mask + online softmax
        #pragma unroll
        for (int r = 0; r < 8; ++r) {
            int row = m0 + lhalf * 8 + r;
            int k0i = kk + l15;
            int k1i = kk + 16 + l15;
            float a = (k0i <= row && row - k0i < WIN) ? sc[0][r] * scale : -3e38f;
            float b = (k1i <= row && row - k1i < WIN) ? sc[1][r] * scale : -3e38f;
            float mx = fmaxf(a, b);
            mx = fmaxf(mx, __shfl_xor(mx, 1, 32));
            mx = fmaxf(mx, __shfl_xor(mx, 2, 32));
            mx = fmaxf(mx, __shfl_xor(mx, 4, 32));
            mx = fmaxf(mx, __shfl_xor(mx, 8, 32));

            float mnew = fmaxf(mrow[r], mx);
            float corr = __expf(mrow[r] - mnew);
            mrow[r] = mnew;
            float p0 = (a > -1e37f) ? __expf(a - mnew) : 0.f;
            float p1 = (b > -1e37f) ? __expf(b - mnew) : 0.f;
            float rs = p0 + p1;
            rs += __shfl_xor(rs, 1, 32);
            rs += __shfl_xor(rs, 2, 32);
            rs += __shfl_xor(rs, 4, 32);
            rs += __shfl_xor(rs, 8, 32);
            lrow[r] = lrow[r] * corr + rs;
            #pragma unroll
            for (int t = 0; t < 8; ++t) acc[t][r] *= corr;

            int rl = lhalf * 8 + r;
            Ps[wave][rl * 40 + l15]      = f2b(p0);
            Ps[wave][rl * 40 + 16 + l15] = f2b(p1);
        }

        // wave-local LDS transpose fence (writes above / reads below, same wave)
        asm volatile("s_wait_dscnt 0" ::: "memory");

        v16bf pfrag;
        {
            const bf16* p = &Ps[wave][l15 * 40 + lhalf * 8];
            uint4* u = (uint4*)&pfrag;
            u[0] = *(const uint4*)p;
            u[1] = *(const uint4*)(p + 16);
        }

        #pragma unroll
        for (int t = 0; t < 8; ++t) {
            v16bf vfrag;
            const bf16* p = Vs + (t * 16 + l15) * 40 + lhalf * 8;
            uint4* u = (uint4*)&vfrag;
            u[0] = *(const uint4*)p;
            u[1] = *(const uint4*)(p + 16);
            acc[t] = __builtin_amdgcn_wmma_f32_16x16x32_bf16(
                false, pfrag, false, vfrag, (short)0, acc[t], false, false);
        }
    }

    #pragma unroll
    for (int t = 0; t < 8; ++t) {
        #pragma unroll
        for (int r = 0; r < 8; ++r) {
            int row = m0 + lhalf * 8 + r;
            int d   = t * 16 + l15;
            O[((size_t)row * NH + h) * HD + d] = f2b(acc[t][r] / lrow[r]);
        }
    }
}

// ---------------------------------------------------------------------------
// Launch
// ---------------------------------------------------------------------------
extern "C" void kernel_launch(void* const* d_in, const int* in_sizes, int n_in,
                              void* d_out, int out_size, void* d_ws, size_t ws_size,
                              hipStream_t stream) {
    (void)in_sizes; (void)n_in; (void)out_size; (void)ws_size;
    const float* x    = (const float*)d_in[0];
    const float* cosb = (const float*)d_in[1];
    const float* sinb = (const float*)d_in[2];
    // d_in[3] (mask) unused: derived analytically in-kernel
    const float* wq   = (const float*)d_in[4];
    const float* wk   = (const float*)d_in[5];
    const float* wv   = (const float*)d_in[6];
    const float* wo   = (const float*)d_in[7];
    float* out = (float*)d_out;

    char* p = (char*)d_ws;
    auto carve = [&](size_t bytes) -> char* {
        char* r = p; p += (bytes + 255) & ~(size_t)255; return r;
    };
    bf16*  xb   = (bf16*)carve((size_t)S_TOT * DIM * 2);
    bf16*  wqT  = (bf16*)carve((size_t)DIM * QCOLS * 2);   // [QCOLS][DIM]
    bf16*  wkT  = (bf16*)carve((size_t)DIM * KVCOLS * 2);  // [KVCOLS][DIM]
    bf16*  wvT  = (bf16*)carve((size_t)DIM * KVCOLS * 2);  // [KVCOLS][DIM]
    bf16*  woT  = (bf16*)carve((size_t)QCOLS * DIM * 2);   // [DIM][QCOLS]
    float* Qf   = (float*)carve((size_t)S_TOT * QCOLS * 4);
    float* Kf   = (float*)carve((size_t)S_TOT * KVCOLS * 4);
    float* Vf   = (float*)carve((size_t)S_TOT * KVCOLS * 4);
    bf16*  Qb   = (bf16*)carve((size_t)S_TOT * QCOLS * 2);
    bf16*  Kb   = (bf16*)carve((size_t)S_TOT * KVCOLS * 2);
    bf16*  VtB  = (bf16*)carve((size_t)KVCOLS * S_TOT * 2); // [KVCOLS][S]
    bf16*  AOb  = (bf16*)carve((size_t)S_TOT * QCOLS * 2);

    // bf16 conversions: x layout-preserving, weights transposed to [N][K]
    cvt_f32_bf16<<<(S_TOT * DIM + 255) / 256, 256, 0, stream>>>(x, xb, S_TOT * DIM);
    cvt_transpose_bf16<<<dim3(QCOLS / 32,  DIM / 32),   256, 0, stream>>>(wq, wqT, DIM,   QCOLS);
    cvt_transpose_bf16<<<dim3(KVCOLS / 32, DIM / 32),   256, 0, stream>>>(wk, wkT, DIM,   KVCOLS);
    cvt_transpose_bf16<<<dim3(KVCOLS / 32, DIM / 32),   256, 0, stream>>>(wv, wvT, DIM,   KVCOLS);
    cvt_transpose_bf16<<<dim3(DIM / 32,    QCOLS / 32), 256, 0, stream>>>(wo, woT, QCOLS, DIM);

    // projections
    gemm_bf16_wmma<<<dim3(QCOLS / 128,  S_TOT / 128), 256, 0, stream>>>(xb, wqT, Qf, S_TOT, QCOLS,  DIM);
    gemm_bf16_wmma<<<dim3(KVCOLS / 128, S_TOT / 128), 256, 0, stream>>>(xb, wkT, Kf, S_TOT, KVCOLS, DIM);
    gemm_bf16_wmma<<<dim3(KVCOLS / 128, S_TOT / 128), 256, 0, stream>>>(xb, wvT, Vf, S_TOT, KVCOLS, DIM);

    // RoPE(Q), RoPE(K); V transposed to [KVCOLS][S] bf16
    rope_bf16<<<(S_TOT * NH  * 64 + 255) / 256, 256, 0, stream>>>(Qf, cosb, sinb, Qb, NH);
    rope_bf16<<<(S_TOT * NKV * 64 + 255) / 256, 256, 0, stream>>>(Kf, cosb, sinb, Kb, NKV);
    cvt_transpose_bf16<<<dim3(KVCOLS / 32, S_TOT / 32), 256, 0, stream>>>(Vf, VtB, S_TOT, KVCOLS);

    // flash attention
    attn_flash_wmma<<<dim3(NH, S_TOT / 128), 256, 0, stream>>>(Qb, Kb, VtB, AOb);

    // output projection -> f32 d_out
    gemm_bf16_wmma<<<dim3(DIM / 128, S_TOT / 128), 256, 0, stream>>>(AOb, woT, out, S_TOT, DIM, QCOLS);
}
